// GroupedQueryAttention_12592844112130
// MI455X (gfx1250) — matrix operand order repaired and checked by reading, not compiled, not run on hardware
//
#include <hip/hip_runtime.h>
#include <hip/hip_bf16.h>
#include <stdint.h>
#include <stddef.h>

// ---------------------------------------------------------------------------
// GQA block for MI455X (gfx1250, wave32).
// ~56 GFLOP vs ~45MB of traffic (fits in 192MB L2; HBM time ~2us @23.3TB/s)
// -> strongly compute bound -> all matmuls on v_wmma_f32_16x16x32_bf16.
// GEMM tiles are double-buffered through LDS via GLOBAL_LOAD_ASYNC_TO_LDS
// (ASYNCcnt path, toolchain-confirmed builtin).
// ---------------------------------------------------------------------------

#define DIMC   1024
#define NHEADS 16
#define NKV    4
#define HD     64
#define SEQ    2048
#define BATCH  2
#define MTOT   (BATCH * SEQ)     // 4096 rows for all projections
#define SCALEF 0.125f            // 64^-0.5

typedef __bf16 bf16_t;
typedef __attribute__((ext_vector_type(16))) __bf16           v16bf;
typedef __attribute__((ext_vector_type(8)))  __bf16           v8bf;
typedef __attribute__((ext_vector_type(8)))  float            v8f;
typedef __attribute__((ext_vector_type(16))) unsigned short   v16us;
typedef __attribute__((ext_vector_type(8)))  unsigned short   v8us;
typedef __attribute__((ext_vector_type(4)))  unsigned short   v4us;
typedef __attribute__((ext_vector_type(4)))  int              v4i;

#define AS_GLOBAL __attribute__((address_space(1)))
#define AS_LOCAL  __attribute__((address_space(3)))

#if defined(__has_builtin)
#  if __has_builtin(__builtin_amdgcn_global_load_async_to_lds_b128)
#    define HAS_ASYNC_LDS 1
#  else
#    define HAS_ASYNC_LDS 0
#  endif
#else
#  define HAS_ASYNC_LDS 0
#endif

// round-to-nearest-even f32 -> bf16 (as raw ushort)
static __device__ __forceinline__ unsigned short f2bf(float f) {
  union { float f; unsigned u; } x; x.f = f;
  unsigned r = x.u + 0x7FFFu + ((x.u >> 16) & 1u);
  return (unsigned short)(r >> 16);
}

static __device__ __forceinline__ v16bf as_v16bf(v16us x) {
  return __builtin_bit_cast(v16bf, x);
}
static __device__ __forceinline__ v16bf join_v16bf(v8us lo, v8us hi) {
  v8bf l = __builtin_bit_cast(v8bf, lo);
  v8bf h = __builtin_bit_cast(v8bf, hi);
  return __builtin_shufflevector(l, h, 0,1,2,3,4,5,6,7,8,9,10,11,12,13,14,15);
}

// 16B global -> LDS copy: async (ASYNCcnt) when available, else via VGPRs.
static __device__ __forceinline__ void cp16_g2l(unsigned short* lds,
                                                const unsigned short* g) {
#if HAS_ASYNC_LDS
  __builtin_amdgcn_global_load_async_to_lds_b128(
      (AS_GLOBAL v4i*)g, (AS_LOCAL v4i*)lds, 0, 0);
#else
  *(v8us*)lds = *(const v8us*)g;
#endif
}
static __device__ __forceinline__ void async_wait_all() {
#if HAS_ASYNC_LDS
#  if __has_builtin(__builtin_amdgcn_s_wait_asynccnt)
  __builtin_amdgcn_s_wait_asynccnt(0);
#  else
  asm volatile("s_wait_asynccnt 0" ::: "memory");
#  endif
#endif
}

// ---------------------------------------------------------------------------
// Pre-pass 1: f32 -> bf16 cast (vectorized x4)
// ---------------------------------------------------------------------------
__global__ void cast_to_bf16_k(const float* __restrict__ in,
                               unsigned short* __restrict__ out, int n4) {
  int i = blockIdx.x * blockDim.x + threadIdx.x;
  if (i < n4) {
    float4 f = ((const float4*)in)[i];
    v4us o;
    o[0] = f2bf(f.x); o[1] = f2bf(f.y); o[2] = f2bf(f.z); o[3] = f2bf(f.w);
    ((v4us*)out)[i] = o;
  }
}

// ---------------------------------------------------------------------------
// Pre-pass 2: W (K x N, f32) -> Wt (N x K, bf16), LDS-tiled 32x32 transpose.
// ---------------------------------------------------------------------------
__global__ void transpose_w_bf16(const float* __restrict__ W,
                                 unsigned short* __restrict__ Wt,
                                 int K, int N) {
  __shared__ float t[32][33];
  const int n0 = blockIdx.x * 32, k0 = blockIdx.y * 32;
  const int x = threadIdx.x, y = threadIdx.y;   // block (32,8)
  #pragma unroll
  for (int i = 0; i < 32; i += 8)
    t[y + i][x] = W[(size_t)(k0 + y + i) * N + n0 + x];
  __syncthreads();
  #pragma unroll
  for (int i = 0; i < 32; i += 8)
    Wt[(size_t)(n0 + y + i) * K + k0 + x] = f2bf(t[x][y + i]);
}

// ---------------------------------------------------------------------------
// Tiled WMMA GEMM: C(MxN) = A(MxK,bf16) * Bt(NxK,bf16)^T
// 256 threads = 8 waves; block tile 128x128; wave tile 32x64; K-step 32.
// LDS double-buffered; staging via async global->LDS.
// OMODE: 0 = f32 out + bias, 1 = bf16 out, 2 = bf16 out transposed (N x M).
// ---------------------------------------------------------------------------
template <int OMODE>
__global__ __launch_bounds__(256)
void gemm_bf16(const unsigned short* __restrict__ A,
               const unsigned short* __restrict__ Bt,
               void* __restrict__ Cout,
               const float* __restrict__ bias,
               int M, int N, int K) {
  __shared__ __align__(32) unsigned short As[2][128 * 32];
  __shared__ __align__(32) unsigned short Bs[2][128 * 32];

  const int tid  = threadIdx.x;
  const int lane = tid & 31;
  const int wave = tid >> 5;              // 0..7
  const int wm = wave >> 1;               // 0..3  (M sub-block of 32)
  const int wn = wave & 1;                // 0..1  (N sub-block of 64)
  const int lm = lane & 15, lh = lane >> 4;
  const int m0 = blockIdx.y * 128;
  const int n0 = blockIdx.x * 128;

  // cooperative loader: thread -> (row 0..127, 16-elem chunk 0/16)
  const int ldRow = tid >> 1;
  const int ldCol = (tid & 1) * 16;
  const unsigned short* gA = &A[(size_t)(m0 + ldRow) * K + ldCol];
  const unsigned short* gB = &Bt[(size_t)(n0 + ldRow) * K + ldCol];

  v8f acc[2][4] = {};

  // stage K-slice k0 into LDS buffer p (two b128 per thread per matrix)
  auto stage = [&](int p, int k0) {
    unsigned short* la = &As[p][ldRow * 32 + ldCol];
    unsigned short* lb = &Bs[p][ldRow * 32 + ldCol];
    cp16_g2l(la,     gA + k0);
    cp16_g2l(la + 8, gA + k0 + 8);
    cp16_g2l(lb,     gB + k0);
    cp16_g2l(lb + 8, gB + k0 + 8);
  };

  stage(0, 0);
  async_wait_all();
  __syncthreads();

  int p = 0;
  for (int k0 = 0; k0 < K; k0 += 32, p ^= 1) {
    if (k0 + 32 < K) stage(p ^ 1, k0 + 32);   // overlap next-tile staging

    // A-frag (16x32 bf16): lane m=lm, elems j -> k = 16*lh + j (contiguous)
    v16bf af[2], bfrag[4];
    #pragma unroll
    for (int i = 0; i < 2; ++i) {
      const int row = wm * 32 + i * 16 + lm;
      af[i] = as_v16bf(*(const v16us*)&As[p][row * 32 + 16 * lh]);
    }
    // B-frag (32x16 bf16): lane n=lm, elems j -> k = 16*lh + j (contiguous)
    #pragma unroll
    for (int j = 0; j < 4; ++j) {
      const int row = wn * 64 + j * 16 + lm;
      bfrag[j] = as_v16bf(*(const v16us*)&Bs[p][row * 32 + 16 * lh]);
    }
    #pragma unroll
    for (int i = 0; i < 2; ++i)
      #pragma unroll
      for (int j = 0; j < 4; ++j)
        acc[i][j] = __builtin_amdgcn_wmma_f32_16x16x32_bf16(
            false, af[i], false, bfrag[j], (short)0, acc[i][j], false, false);

    async_wait_all();   // next buffer's async copies have landed
    __syncthreads();
  }

  // epilogue: C layout lane -> (m = r + 8*lh, n = lm) within each 16x16 tile
  #pragma unroll
  for (int i = 0; i < 2; ++i) {
    #pragma unroll
    for (int j = 0; j < 4; ++j) {
      const int nn = n0 + wn * 64 + j * 16 + lm;
      const float bb = (OMODE == 0) ? bias[nn] : 0.0f;
      #pragma unroll
      for (int r = 0; r < 8; ++r) {
        const int mm = m0 + wm * 32 + i * 16 + r + 8 * lh;
        const float v = acc[i][j][r];
        if (OMODE == 0)
          ((float*)Cout)[(size_t)mm * N + nn] = v + bb;
        else if (OMODE == 1)
          ((unsigned short*)Cout)[(size_t)mm * N + nn] = f2bf(v);
        else
          ((unsigned short*)Cout)[(size_t)nn * M + mm] = f2bf(v);
      }
    }
  }
}

// ---------------------------------------------------------------------------
// Flash-style GQA attention. One wave owns 16 queries of one head; online
// softmax over 64-key blocks. Q/K/P/V matmuls all via WMMA bf16.
//   qb : (B*SEQ, 1024) bf16   kb : (B*SEQ, 256) bf16
//   vt : (256, B*SEQ) bf16    ob : (B*SEQ, 1024) bf16
// ---------------------------------------------------------------------------
__global__ __launch_bounds__(128)
void gqa_attention(const unsigned short* __restrict__ qb,
                   const unsigned short* __restrict__ kb,
                   const unsigned short* __restrict__ vt,
                   unsigned short* __restrict__ ob) {
  __shared__ __align__(32) unsigned short Pl[4][16 * 64]; // per-wave P buffer

  const int lane = threadIdx.x & 31;
  const int wave = threadIdx.x >> 5;
  const int lm = lane & 15, lh = lane >> 4;

  const int gw   = blockIdx.x * 4 + wave;   // 0..4095 query blocks
  const int qblk = gw & 127;                // SEQ/16 = 128
  const int head = (gw >> 7) & 15;
  const int b    = gw >> 11;
  const int kvh  = head >> 2;               // GROUP = 4
  const int q0   = qblk * 16;

  // resident Q fragments (16 queries x 64 dims -> two 16x32 A-frags)
  const unsigned short* qp =
      qb + ((size_t)(b * SEQ + q0 + lm) * DIMC + head * HD);
  v16bf qf[2];
  #pragma unroll
  for (int t = 0; t < 2; ++t) {
    qf[t] = as_v16bf(*(const v16us*)&qp[t * 32 + 16 * lh]);
  }

  const unsigned short* kbase = kb + ((size_t)(b * SEQ) * (NKV * HD) + kvh * HD);
  const unsigned short* vbase = vt + ((size_t)(kvh * HD) * MTOT + b * SEQ);
  unsigned short* Pw = &Pl[wave][0];

  v8f o[4] = {};                 // 16 queries x 64 dims accumulator
  float mrow[8], lrow[8];        // per-row (m = r + 8*lh) online stats
  #pragma unroll
  for (int r = 0; r < 8; ++r) { mrow[r] = -1e30f; lrow[r] = 0.0f; }

  for (int kb0 = 0; kb0 < SEQ; kb0 += 64) {
    // prefetch next key block of K and V^T (global_prefetch_b8)
    if (kb0 + 64 < SEQ) {
      #pragma unroll
      for (int r = 0; r < 2; ++r) {
        __builtin_prefetch(kbase + (size_t)(kb0 + 64 + lane * 2 + r) * (NKV * HD), 0, 3);
        __builtin_prefetch(vbase + (size_t)(lane * 2 + r) * MTOT + kb0 + 64, 0, 3);
      }
    }

    // ---- batch-load all 8 K fragments, then 8 score WMMAs ----
    v16bf kf[4][2];
    #pragma unroll
    for (int kt = 0; kt < 4; ++kt) {
      const unsigned short* kp = kbase + (size_t)(kb0 + kt * 16 + lm) * (NKV * HD);
      #pragma unroll
      for (int t = 0; t < 2; ++t)
        kf[kt][t] = as_v16bf(*(const v16us*)&kp[t * 32 + 16 * lh]);
    }
    v8f s[4] = {};
    #pragma unroll
    for (int kt = 0; kt < 4; ++kt)
      #pragma unroll
      for (int t = 0; t < 2; ++t)
        s[kt] = __builtin_amdgcn_wmma_f32_16x16x32_bf16(
            false, qf[t], false, kf[kt][t], (short)0, s[kt], false, false);

    // ---- issue V-fragment loads early: independent of softmax VALU ----
    v16bf vf[4][2];
    #pragma unroll
    for (int t = 0; t < 4; ++t) {
      const unsigned short* vp = vbase + (size_t)(t * 16 + lm) * MTOT + kb0;
      #pragma unroll
      for (int u = 0; u < 2; ++u)
        vf[t][u] = as_v16bf(*(const v16us*)&vp[u * 32 + 16 * lh]);
    }

    // ---- online softmax on raw scores; scale folded into exp via fma ----
    #pragma unroll
    for (int r = 0; r < 8; ++r) {
      float bm = fmaxf(fmaxf(s[0][r], s[1][r]), fmaxf(s[2][r], s[3][r]));
      #pragma unroll
      for (int d = 1; d < 16; d <<= 1) bm = fmaxf(bm, __shfl_xor(bm, d, 32));
      const float mn = fmaxf(mrow[r], bm * SCALEF);   // max commutes with +scale
      const float corr = __expf(mrow[r] - mn);
      mrow[r] = mn;
      float rs = 0.0f;
      #pragma unroll
      for (int kt = 0; kt < 4; ++kt) {
        float pv = __expf(__builtin_fmaf(s[kt][r], SCALEF, -mn));
        s[kt][r] = pv;
        rs += pv;
      }
      #pragma unroll
      for (int d = 1; d < 16; d <<= 1) rs += __shfl_xor(rs, d, 32);
      lrow[r] = lrow[r] * corr + rs;
      #pragma unroll
      for (int t = 0; t < 4; ++t) o[t][r] *= corr;
    }

    // ---- P: accumulator layout -> A-frag layout via per-wave LDS ----
    #pragma unroll
    for (int kt = 0; kt < 4; ++kt)
      #pragma unroll
      for (int r = 0; r < 8; ++r)
        Pw[(size_t)(r + 8 * lh) * 64 + kt * 16 + lm] = f2bf(s[kt][r]);

    v16bf pf[2];
    #pragma unroll
    for (int u = 0; u < 2; ++u) {   // two 16x32 A-frags over the 64 keys
      pf[u] = as_v16bf(*(const v16us*)&Pw[(size_t)lm * 64 + u * 32 + 16 * lh]);
    }

    // ---- O += P V : all fragments resident, 8 back-to-back WMMAs ----
    #pragma unroll
    for (int t = 0; t < 4; ++t)
      #pragma unroll
      for (int u = 0; u < 2; ++u)
        o[t] = __builtin_amdgcn_wmma_f32_16x16x32_bf16(
            false, pf[u], false, vf[t][u], (short)0, o[t], false, false);
  }

  // ---- epilogue: O * (1/l) -> bf16 attention output ----
  unsigned short* op = ob + ((size_t)(b * SEQ + q0) * DIMC + head * HD);
  float linv[8];
  #pragma unroll
  for (int r = 0; r < 8; ++r) linv[r] = __builtin_amdgcn_rcpf(lrow[r]);
  #pragma unroll
  for (int t = 0; t < 4; ++t)
    #pragma unroll
    for (int r = 0; r < 8; ++r)
      op[(size_t)(r + 8 * lh) * DIMC + t * 16 + lm] = f2bf(o[t][r] * linv[r]);
}

// ---------------------------------------------------------------------------
// Host: stage pipeline on `stream`. Workspace layout (bytes, ~33 MB total):
//   xb 8M | wqt 2M | wkt .5M | wvt .5M | wot 2M | qb 8M | kb 2M | vt 2M | ao 8M
// ---------------------------------------------------------------------------
extern "C" void kernel_launch(void* const* d_in, const int* in_sizes, int n_in,
                              void* d_out, int out_size, void* d_ws, size_t ws_size,
                              hipStream_t stream) {
  const float* x  = (const float*)d_in[0];
  const float* Wq = (const float*)d_in[1];
  const float* Wk = (const float*)d_in[2];
  const float* Wv = (const float*)d_in[3];
  const float* Wo = (const float*)d_in[4];
  const float* bo = (const float*)d_in[5];
  float* out = (float*)d_out;

  char* ws = (char*)d_ws;
  size_t off = 0;
  unsigned short* xb  = (unsigned short*)(ws + off); off += (size_t)MTOT * DIMC * 2;
  unsigned short* wqt = (unsigned short*)(ws + off); off += (size_t)DIMC * DIMC * 2;
  unsigned short* wkt = (unsigned short*)(ws + off); off += (size_t)(NKV*HD) * DIMC * 2;
  unsigned short* wvt = (unsigned short*)(ws + off); off += (size_t)(NKV*HD) * DIMC * 2;
  unsigned short* wot = (unsigned short*)(ws + off); off += (size_t)DIMC * DIMC * 2;
  unsigned short* qbf = (unsigned short*)(ws + off); off += (size_t)MTOT * DIMC * 2;
  unsigned short* kbf = (unsigned short*)(ws + off); off += (size_t)MTOT * (NKV*HD) * 2;
  unsigned short* vtb = (unsigned short*)(ws + off); off += (size_t)MTOT * (NKV*HD) * 2;
  unsigned short* aob = (unsigned short*)(ws + off); off += (size_t)MTOT * DIMC * 2;

  const dim3 b256(256);
  const dim3 b128(128);
  const dim3 btr(32, 8);

  // 1) casts / weight transposes to bf16
  {
    int n4 = MTOT * DIMC / 4;
    cast_to_bf16_k<<<(n4 + 255) / 256, 256, 0, stream>>>(x, xb, n4);
  }
  transpose_w_bf16<<<dim3(DIMC/32, DIMC/32), btr, 0, stream>>>(Wq, wqt, DIMC, DIMC);
  transpose_w_bf16<<<dim3((NKV*HD)/32, DIMC/32), btr, 0, stream>>>(Wk, wkt, DIMC, NKV*HD);
  transpose_w_bf16<<<dim3((NKV*HD)/32, DIMC/32), btr, 0, stream>>>(Wv, wvt, DIMC, NKV*HD);
  transpose_w_bf16<<<dim3(DIMC/32, DIMC/32), btr, 0, stream>>>(Wo, wot, DIMC, DIMC);

  // 2) projections (WMMA GEMMs, 128x128 block tiles)
  gemm_bf16<1><<<dim3(DIMC/128,     MTOT/128), b256, 0, stream>>>(xb, wqt, qbf, nullptr, MTOT, DIMC,   DIMC);
  gemm_bf16<1><<<dim3((NKV*HD)/128, MTOT/128), b256, 0, stream>>>(xb, wkt, kbf, nullptr, MTOT, NKV*HD, DIMC);
  gemm_bf16<2><<<dim3((NKV*HD)/128, MTOT/128), b256, 0, stream>>>(xb, wvt, vtb, nullptr, MTOT, NKV*HD, DIMC);

  // 3) attention: 2 batches * 16 heads * 128 query blocks = 4096 waves
  gqa_attention<<<dim3((BATCH * NHEADS * (SEQ/16)) / 4), b128, 0, stream>>>(qbf, kbf, vtb, aob);

  // 4) output projection + bias (f32 out)
  gemm_bf16<0><<<dim3(DIMC/128, MTOT/128), b256, 0, stream>>>(aob, wot, out, bo, MTOT, DIMC, DIMC);
}
